// RegionBlock_49821620633767
// MI455X (gfx1250) — matrix operand order repaired and checked by reading, not compiled, not run on hardware
//
#include <hip/hip_runtime.h>
#include <math.h>

#define DEVINL static __device__ __forceinline__

typedef __attribute__((ext_vector_type(16))) __bf16 v16bf;
typedef __attribute__((ext_vector_type(8)))  float  v8f;
typedef __attribute__((ext_vector_type(4)))  unsigned int v4u;
typedef __attribute__((ext_vector_type(8)))  int v8i;
typedef __attribute__((ext_vector_type(4)))  int v4i;

// Address-space-qualified pointer types (attribute must live in a typedef to stick)
typedef __attribute__((address_space(1))) v4i* gv4p;  // global int4*
typedef __attribute__((address_space(3))) v4i* lv4p;  // LDS int4*

// ---- CDNA5 async / TDM feature detection (graceful fallback) --------------
#if __has_builtin(__builtin_amdgcn_global_load_async_to_lds_b128)
#define RB_ASYNC 1
#else
#define RB_ASYNC 0
#endif

#if __has_builtin(__builtin_amdgcn_s_wait_asynccnt)
#define RB_WAIT_ASYNC() __builtin_amdgcn_s_wait_asynccnt(0)
#else
#define RB_WAIT_ASYNC() asm volatile("s_wait_asynccnt 0x0" ::: "memory")
#endif

#if __has_builtin(__builtin_amdgcn_tensor_load_to_lds)
#define RB_TDM 1
#else
#define RB_TDM 0
#endif

#if __has_builtin(__builtin_amdgcn_s_wait_tensorcnt)
#define RB_WAIT_TENSOR() __builtin_amdgcn_s_wait_tensorcnt(0)
#else
#define RB_WAIT_TENSOR() asm volatile("s_wait_tensorcnt 0x0" ::: "memory")
#endif

DEVINL void asyncLoad16(const void* gsrc, void* ldst) {
#if RB_ASYNC
  __builtin_amdgcn_global_load_async_to_lds_b128((gv4p)gsrc, (lv4p)ldst, 0, 0);
#else
  (void)gsrc; (void)ldst;
#endif
}

namespace rb {
constexpr int B = 8, N = 9237, C = 512, T = 21, H = 8, D = 64, HID = 2048;
constexpr int TOTROWS = B * N;                 // 73896
constexpr float EPS = 1e-5f;
constexpr int NTCH = (N + 255) / 256;          // 37 chunks for final scores
constexpr int NSPLIT = 8;                      // blocks per (b,h) in attn0
constexpr int CH = (N + NSPLIT - 1) / NSPLIT;  // 1155
constexpr int VSTR = 272;                      // padded LDS stride for v^T tile
}
static_assert(rb::CH * rb::NSPLIT >= rb::N, "split covers N");

union V16BF { v16bf v; unsigned short s[16]; uint4 q[2]; };

DEVINL unsigned short f2bf(float f) {
  unsigned u = __float_as_uint(f);
  return (unsigned short)((u + 0x7fffu + ((u >> 16) & 1u)) >> 16);
}

DEVINL v8f wmma_bf16(v16bf a, v16bf b, v8f c) {
  // D = A(16x32 bf16) * B(32x16 bf16) + C(16x16 f32)
  return __builtin_amdgcn_wmma_f32_16x16x32_bf16(false, a, false, b, (short)0, c, false, false);
}

DEVINL float redSum32(float v) { for (int m = 1; m < 32; m <<= 1) v += __shfl_xor(v, m, 32); return v; }
DEVINL float redSum16(float v) { for (int m = 1; m < 16; m <<= 1) v += __shfl_xor(v, m, 32); return v; }
DEVINL float redMax16(float v) { for (int m = 1; m < 16; m <<= 1) v = fmaxf(v, __shfl_xor(v, m, 32)); return v; }

// A fragment (16x32 bf16): lane row = lane%16; lanes<16 hold K {kb+0..7, kb+16..23},
// lanes>=16 hold K {kb+8..15, kb+24..31}.  Source row-major, 'stride' elements/row.
DEVINL v16bf aFrag(const unsigned short* base, int row, int stride, int half, int kb) {
  const unsigned short* p = base + row * stride + kb + half * 8;
  V16BF t; t.q[0] = *(const uint4*)p; t.q[1] = *(const uint4*)(p + 16); return t.v;
}
// B fragment (32x16 bf16): lane col = lane%16; lanes<16 hold K {kb+0..7, kb+16..23}, lanes>=16 K {kb+8..15, kb+24..31}.
// Source stored K-contiguous per column ('row' = column index, 'stride' elems per column).
DEVINL v16bf bFrag(const unsigned short* base, size_t row, int stride, int half, int kb) {
  const unsigned short* p = base + row * stride + kb + half * 8;
  V16BF t; t.q[0] = *(const uint4*)p; t.q[1] = *(const uint4*)(p + 16); return t.v;
}
// Build A fragment from an f32 query row (64 contiguous K values), converting to bf16.
DEVINL v16bf qFrag(const float* qrow, int half, int ks) {
  V16BF t;
#pragma unroll
  for (int j = 0; j < 16; ++j) {
    int K = ks * 32 + ((j < 8) ? (half * 8 + j) : (16 + half * 8 + (j - 8)));
    t.s[j] = f2bf(qrow[K]);
  }
  return t.v;
}

// ---------------------------------------------------------------------------
// K0: fold layernorm gains into transposed bf16 weights, biases = b . W
// ---------------------------------------------------------------------------
__global__ __launch_bounds__(128) void rb_prep_w(
    const float* __restrict__ kv0w, const float* __restrict__ kv1w,
    const float* __restrict__ g1, const float* __restrict__ b1,
    const float* __restrict__ g3, const float* __restrict__ b3,
    unsigned short* __restrict__ wt0, float* __restrict__ bias0,
    unsigned short* __restrict__ wt1, float* __restrict__ bias1) {
  using namespace rb;
  __shared__ float red[128];
  const int col = blockIdx.x, tid = threadIdx.x;
  const float *W, *g, *bv; unsigned short* wo; float* bo; int c;
  if (col < 1024) { W = kv0w; g = g1; bv = b1; c = col;        wo = wt0 + (size_t)c * C; bo = bias0 + c; }
  else            { W = kv1w; g = g3; bv = b3; c = col - 1024; wo = wt1 + (size_t)c * C; bo = bias1 + c; }
  float part = 0.f;
  for (int k2 = tid; k2 < C; k2 += 128) {
    float wv = W[(long)k2 * 1024 + c];
    wo[k2] = f2bf(g[k2] * wv);
    part += bv[k2] * wv;
  }
  red[tid] = part; __syncthreads();
  for (int s2 = 64; s2 > 0; s2 >>= 1) { if (tid < s2) red[tid] += red[tid + s2]; __syncthreads(); }
  if (tid == 0) *bo = red[0];
}

// ---------------------------------------------------------------------------
// K1: fused layernorm (xhat) + dual GEMM -> k0,v0 (1024 cols) and k1 (512 cols).
// Block = 128 rows x full K=512 in LDS; 8 waves, each owns a 16-row strip.
// Column tiles processed in pairs: one A-frag feeds two WMMA chains.
// ---------------------------------------------------------------------------
__global__ __launch_bounds__(256) void rb_ln_dualgemm(
    const float* __restrict__ x,
    const unsigned short* __restrict__ wt0, const float* __restrict__ bias0,
    const unsigned short* __restrict__ wt1, const float* __restrict__ bias1,
    unsigned short* __restrict__ k0, unsigned short* __restrict__ v0,
    unsigned short* __restrict__ k1) {
  using namespace rb;
  __shared__ unsigned short xh[128 * C];  // 128 KB xhat tile (bf16)
  const int tid = threadIdx.x, lane = tid & 31, w = tid >> 5;
  const int half = lane >> 4, nn = lane & 15;
  const long r0 = (long)blockIdx.x * 128;
  const bool fullBlk = (r0 + 128 <= TOTROWS);

  // Phase A: per-row mean/var + xhat -> LDS (wave cooperates on one row at a time)
  for (int rr = 0; rr < 16; ++rr) {
    long row = r0 + w * 16 + rr;
    float4 vv[4] = {};
    if (row < TOTROWS) {
      const float4* xp = (const float4*)(x + row * C);
#pragma unroll
      for (int j = 0; j < 4; ++j) vv[j] = xp[j * 32 + lane];
    }
    float s = 0.f;
#pragma unroll
    for (int j = 0; j < 4; ++j) s += vv[j].x + vv[j].y + vv[j].z + vv[j].w;
    s = redSum32(s);
    float mean = s * (1.f / C);
    float q2 = 0.f;
#pragma unroll
    for (int j = 0; j < 4; ++j) {
      float a = vv[j].x - mean, b2 = vv[j].y - mean, c2 = vv[j].z - mean, d2 = vv[j].w - mean;
      q2 += a * a + b2 * b2 + c2 * c2 + d2 * d2;
    }
    q2 = redSum32(q2);
    float rs = rsqrtf(q2 * (1.f / C) + EPS);
    unsigned short* orow = xh + (w * 16 + rr) * C;
#pragma unroll
    for (int j = 0; j < 4; ++j) {
      int c0 = (j * 32 + lane) * 4;
      orow[c0 + 0] = f2bf((vv[j].x - mean) * rs);
      orow[c0 + 1] = f2bf((vv[j].y - mean) * rs);
      orow[c0 + 2] = f2bf((vv[j].z - mean) * rs);
      orow[c0 + 3] = f2bf((vv[j].w - mean) * rs);
    }
  }
  __syncthreads();

  // Destination base per output row of this lane's C-fragment rows (M = r + 8*half)
  long bases[8];
#pragma unroll
  for (int r = 0; r < 8; ++r) {
    long row = r0 + w * 16 + r + 8 * half;
    if (row < TOTROWS) {
      long bb = row / N, nidx = row - bb * N;
      bases[r] = (bb * H) * (long)N * 64 + nidx * 64;
    } else bases[r] = -1;
  }
  const int strip = w * 16;

  auto storeKV0 = [&](v8f acc, int c) {
    float bia = bias0[c];
    int kvsel = c >> 9, hh = (c >> 6) & 7, dd = c & 63;
    unsigned short* dst = kvsel ? v0 : k0;
    long hoff = (long)hh * N * 64 + dd;
    if (fullBlk) {
#pragma unroll
      for (int r = 0; r < 8; ++r) dst[bases[r] + hoff] = f2bf(acc[r] + bia);
    } else {
#pragma unroll
      for (int r = 0; r < 8; ++r)
        if (bases[r] >= 0) dst[bases[r] + hoff] = f2bf(acc[r] + bia);
    }
  };
  auto storeK1 = [&](v8f acc, int c) {
    float bia = bias1[c];
    int hh = c >> 6, dd = c & 63;
    long hoff = (long)hh * N * 64 + dd;
    if (fullBlk) {
#pragma unroll
      for (int r = 0; r < 8; ++r) k1[bases[r] + hoff] = f2bf(acc[r] + bia);
    } else {
#pragma unroll
      for (int r = 0; r < 8; ++r)
        if (bases[r] >= 0) k1[bases[r] + hoff] = f2bf(acc[r] + bia);
    }
  };

  // Phase B: xhat @ wt0 -> k0/v0 (64 col tiles, pairs)
  for (int ct = 0; ct < 64; ct += 2) {
    v8f acc0 = {0.f, 0.f, 0.f, 0.f, 0.f, 0.f, 0.f, 0.f};
    v8f acc1 = {0.f, 0.f, 0.f, 0.f, 0.f, 0.f, 0.f, 0.f};
#pragma unroll 4
    for (int kt = 0; kt < 16; ++kt) {
      v16bf a = aFrag(xh, strip + nn, C, half, kt * 32);
      v16bf b0 = bFrag(wt0, (size_t)(ct * 16 + nn), C, half, kt * 32);
      v16bf b1 = bFrag(wt0, (size_t)((ct + 1) * 16 + nn), C, half, kt * 32);
      acc0 = wmma_bf16(a, b0, acc0);
      acc1 = wmma_bf16(a, b1, acc1);
    }
    storeKV0(acc0, ct * 16 + nn);
    storeKV0(acc1, (ct + 1) * 16 + nn);
  }
  // Phase C: xhat @ wt1 -> k1 (32 col tiles, pairs)
  for (int ct = 0; ct < 32; ct += 2) {
    v8f acc0 = {0.f, 0.f, 0.f, 0.f, 0.f, 0.f, 0.f, 0.f};
    v8f acc1 = {0.f, 0.f, 0.f, 0.f, 0.f, 0.f, 0.f, 0.f};
#pragma unroll 4
    for (int kt = 0; kt < 16; ++kt) {
      v16bf a = aFrag(xh, strip + nn, C, half, kt * 32);
      v16bf b0 = bFrag(wt1, (size_t)(ct * 16 + nn), C, half, kt * 32);
      v16bf b1 = bFrag(wt1, (size_t)((ct + 1) * 16 + nn), C, half, kt * 32);
      acc0 = wmma_bf16(a, b0, acc0);
      acc1 = wmma_bf16(a, b1, acc1);
    }
    storeK1(acc0, ct * 16 + nn);
    storeK1(acc1, (ct + 1) * 16 + nn);
  }
}

// ---------------------------------------------------------------------------
// K2: q = (ln(src[:, :T]) @ qw) with the non-standard [B,T,H*D]->[B,H,T,D] reshape,
// scaled by D^-0.5, stored padded [B,H,32,64] f32 (rows T..31 zeroed).
// ---------------------------------------------------------------------------
__global__ __launch_bounds__(256) void rb_compute_q(
    const float* __restrict__ src, long batchStride,
    const float* __restrict__ g, const float* __restrict__ bb,
    const float* __restrict__ qw, float* __restrict__ qout, float scale) {
  using namespace rb;
  __shared__ float lnr[T * C];  // 43 KB
  const int tid = threadIdx.x, lane = tid & 31, w = tid >> 5;
  const int b = blockIdx.x;
  for (int t = w; t < T; t += 8) {
    const float4* s4 = (const float4*)(src + (long)b * batchStride + (long)t * C);
    float4 vv[4];
#pragma unroll
    for (int j = 0; j < 4; ++j) vv[j] = s4[j * 32 + lane];
    float s = 0.f;
#pragma unroll
    for (int j = 0; j < 4; ++j) s += vv[j].x + vv[j].y + vv[j].z + vv[j].w;
    s = redSum32(s);
    float mean = s * (1.f / C);
    float q2 = 0.f;
#pragma unroll
    for (int j = 0; j < 4; ++j) {
      float a = vv[j].x - mean, b2 = vv[j].y - mean, c2 = vv[j].z - mean, d2 = vv[j].w - mean;
      q2 += a * a + b2 * b2 + c2 * c2 + d2 * d2;
    }
    q2 = redSum32(q2);
    float rs = rsqrtf(q2 * (1.f / C) + EPS);
#pragma unroll
    for (int j = 0; j < 4; ++j) {
      int c0 = (j * 32 + lane) * 4;
      lnr[t * C + c0 + 0] = (vv[j].x - mean) * rs * g[c0 + 0] + bb[c0 + 0];
      lnr[t * C + c0 + 1] = (vv[j].y - mean) * rs * g[c0 + 1] + bb[c0 + 1];
      lnr[t * C + c0 + 2] = (vv[j].z - mean) * rs * g[c0 + 2] + bb[c0 + 2];
      lnr[t * C + c0 + 3] = (vv[j].w - mean) * rs * g[c0 + 3] + bb[c0 + 3];
    }
  }
  __syncthreads();
  for (int i = tid; i < T * C; i += 256) {
    int r = i >> 9, c = i & 511;
    float acc = 0.f;
    for (int k2 = 0; k2 < C; ++k2) acc += lnr[r * C + k2] * qw[(long)k2 * C + c];
    int h = i / (T * D);
    int rem = i - h * (T * D);
    int t = rem >> 6, d = rem & 63;
    qout[(((long)b * H + h) * 32 + t) * 64 + d] = scale * acc;
  }
  for (int i = tid; i < H * (32 - T) * D; i += 256) {
    int h = i / ((32 - T) * D);
    int rem = i % ((32 - T) * D);
    int t = T + rem / D, d = rem % D;
    qout[(((long)b * H + h) * 32 + t) * 64 + d] = 0.f;
  }
}

// ---------------------------------------------------------------------------
// K3: attention-0, flash style. Grid = B*H*NSPLIT blocks; each of 8 waves is an
// independent split (64 splits per (b,h)).  k-tile staged via async-to-LDS.
// ---------------------------------------------------------------------------
__global__ __launch_bounds__(256) void rb_attn0(
    const float* __restrict__ q0, const unsigned short* __restrict__ k0,
    const unsigned short* __restrict__ v0,
    float* __restrict__ pO, float* __restrict__ pML) {
  using namespace rb;
  __shared__ unsigned short kt[256 * 64];    // 32 KB  k tile, [n][d]
  __shared__ unsigned short vt[64 * VSTR];   // 34 KB  v tile transposed, [d][n]
  __shared__ unsigned short pl[8 * 1024];    // 16 KB  per-wave P staging
  const int tid = threadIdx.x, lane = tid & 31, w = tid >> 5;
  const int half = lane >> 4, nn = lane & 15;
  const int bh = blockIdx.x >> 3, s = blockIdx.x & 7;
  const int nStart = s * CH;
  const int nHi = min(nStart + CH, N);
  const float NEGINF = -__builtin_inff();

  v16bf qf[2][2];
#pragma unroll
  for (int mt = 0; mt < 2; ++mt)
#pragma unroll
    for (int ks = 0; ks < 2; ++ks)
      qf[mt][ks] = qFrag(q0 + ((long)bh * 32 + mt * 16 + nn) * 64, half, ks);

  float mRun[2][8], lRun[2][8];
  v8f O[2][4];
#pragma unroll
  for (int mt = 0; mt < 2; ++mt) {
#pragma unroll
    for (int r = 0; r < 8; ++r) { mRun[mt][r] = NEGINF; lRun[mt][r] = 0.f; }
#pragma unroll
    for (int dt = 0; dt < 4; ++dt)
#pragma unroll
      for (int r = 0; r < 8; ++r) O[mt][dt][r] = 0.f;
  }

  const unsigned short* k0b = k0 + (long)bh * N * 64;
  const unsigned short* v0b = v0 + (long)bh * N * 64;
  const int nTiles = (nHi - nStart + 255) / 256;

  for (int tI = 0; tI < nTiles; ++tI) {
    const int tBase = nStart + tI * 256;
    {  // stage 256 rows of k (async DMA to LDS) and v (transposed into [d][n])
      const int n = tBase + tid;
      const bool val = n < nHi;
      uint4 z; z.x = z.y = z.z = z.w = 0u;
      uint4* kd = (uint4*)(kt + tid * 64);
      const uint4* kp = (const uint4*)(k0b + (long)n * 64);
      const uint4* vp = (const uint4*)(v0b + (long)n * 64);
#if RB_ASYNC
      if (val) {
#pragma unroll
        for (int jj = 0; jj < 8; ++jj) asyncLoad16(kp + jj, kd + jj);
      } else {
#pragma unroll
        for (int jj = 0; jj < 8; ++jj) kd[jj] = z;
      }
#else
#pragma unroll
      for (int jj = 0; jj < 8; ++jj) kd[jj] = val ? kp[jj] : z;
#endif
#pragma unroll
      for (int jj = 0; jj < 8; ++jj) {
        uint4 vv_ = val ? vp[jj] : z;
        const unsigned short* vs = (const unsigned short*)&vv_;
#pragma unroll
        for (int cc = 0; cc < 8; ++cc) vt[(jj * 8 + cc) * VSTR + tid] = vs[cc];
      }
      if (tBase + 256 + tid < nHi) {  // prefetch next tile rows into L2/L0
        __builtin_prefetch(k0b + (long)(tBase + 256 + tid) * 64, 0, 0);
        __builtin_prefetch(v0b + (long)(tBase + 256 + tid) * 64, 0, 0);
      }
#if RB_ASYNC
      RB_WAIT_ASYNC();
#endif
    }
    __syncthreads();
    const int nOff = w * 32;
    v8f S[2][2];
#pragma unroll
    for (int mt = 0; mt < 2; ++mt)
#pragma unroll
      for (int sub = 0; sub < 2; ++sub)
#pragma unroll
        for (int r = 0; r < 8; ++r) S[mt][sub][r] = 0.f;
#pragma unroll
    for (int sub = 0; sub < 2; ++sub)
#pragma unroll
      for (int ks = 0; ks < 2; ++ks) {
        v16bf bf = bFrag(kt, (size_t)(nOff + sub * 16 + nn), 64, half, ks * 32);
        S[0][sub] = wmma_bf16(qf[0][ks], bf, S[0][sub]);
        S[1][sub] = wmma_bf16(qf[1][ks], bf, S[1][sub]);
      }
#pragma unroll
    for (int sub = 0; sub < 2; ++sub) {
      int nG = tBase + nOff + sub * 16 + nn;
      if (nG >= nHi) {
#pragma unroll
        for (int mt = 0; mt < 2; ++mt)
#pragma unroll
          for (int r = 0; r < 8; ++r) S[mt][sub][r] = NEGINF;
      }
    }
#pragma unroll
    for (int mt = 0; mt < 2; ++mt) {
      float sc[8];
#pragma unroll
      for (int r = 0; r < 8; ++r) {
        float rm = redMax16(fmaxf(S[mt][0][r], S[mt][1][r]));
        float nm = fmaxf(mRun[mt][r], rm);
        sc[r] = (mRun[mt][r] == NEGINF) ? 0.f : __expf(mRun[mt][r] - nm);
        float p0 = (S[mt][0][r] == NEGINF) ? 0.f : __expf(S[mt][0][r] - nm);
        float p1 = (S[mt][1][r] == NEGINF) ? 0.f : __expf(S[mt][1][r] - nm);
        float rsum = redSum16(p0 + p1);
        lRun[mt][r] = lRun[mt][r] * sc[r] + rsum;
        mRun[mt][r] = nm;
        unsigned short* pp = pl + w * 1024 + mt * 512 + (r + 8 * half) * 32;
        pp[nn] = f2bf(p0);
        pp[16 + nn] = f2bf(p1);
      }
#pragma unroll
      for (int dt = 0; dt < 4; ++dt)
#pragma unroll
        for (int r = 0; r < 8; ++r) O[mt][dt][r] *= sc[r];
      v16bf pf = aFrag(pl + w * 1024 + mt * 512, nn, 32, half, 0);
#pragma unroll
      for (int dt = 0; dt < 4; ++dt) {
        v16bf vf = bFrag(vt, (size_t)(dt * 16 + nn), VSTR, half, nOff);
        O[mt][dt] = wmma_bf16(pf, vf, O[mt][dt]);
      }
    }
    __syncthreads();
  }
  const int sIdx = s * 8 + w;
  const long pb = (long)bh * 64 + sIdx;
#pragma unroll
  for (int mt = 0; mt < 2; ++mt)
#pragma unroll
    for (int dt = 0; dt < 4; ++dt)
#pragma unroll
      for (int r = 0; r < 8; ++r) {
        int t = mt * 16 + r + 8 * half, d = dt * 16 + nn;
        pO[pb * 2048 + t * 64 + d] = O[mt][dt][r];
      }
  if (nn == 0) {
#pragma unroll
    for (int mt = 0; mt < 2; ++mt)
#pragma unroll
      for (int r = 0; r < 8; ++r) {
        int t = mt * 16 + r + 8 * half;
        pML[pb * 64 + t * 2] = mRun[mt][r];
        pML[pb * 64 + t * 2 + 1] = lRun[mt][r];
      }
  }
}

// K3b: combine 64 split partials -> ce [B,T,C] f32 (already head-concatenated)
__global__ __launch_bounds__(256) void rb_attn0_combine(
    const float* __restrict__ pO, const float* __restrict__ pML, float* __restrict__ ce) {
  using namespace rb;
  const int bh = blockIdx.x, b = bh >> 3, h = bh & 7, tid = threadIdx.x;
  __shared__ float sL[32];
  __shared__ float sW[64 * 32];
  const float NEGINF = -__builtin_inff();
  if (tid < 32) {
    int t = tid;
    float M = NEGINF;
    for (int s2 = 0; s2 < 64; ++s2) M = fmaxf(M, pML[((long)bh * 64 + s2) * 64 + t * 2]);
    float L = 0.f;
    for (int s2 = 0; s2 < 64; ++s2) {
      float m = pML[((long)bh * 64 + s2) * 64 + t * 2];
      float l = pML[((long)bh * 64 + s2) * 64 + t * 2 + 1];
      float wgt = (m == NEGINF) ? 0.f : __expf(m - M);
      sW[s2 * 32 + t] = wgt;
      L += l * wgt;
    }
    sL[t] = (L > 0.f) ? L : 1.f;
  }
  __syncthreads();
  for (int i = tid; i < T * D; i += 256) {
    int t = i >> 6, d = i & 63;
    float o = 0.f;
    for (int s2 = 0; s2 < 64; ++s2) o += pO[((long)bh * 64 + s2) * 2048 + t * 64 + d] * sW[s2 * 32 + t];
    ce[((long)b * T + t) * C + h * D + d] = o / sL[t];
  }
}

// K4a: cls = x[:, :T] + ce @ proj0_w + proj0_b
__global__ __launch_bounds__(256) void rb_proj_res(
    const float* __restrict__ ce, const float* __restrict__ pw, const float* __restrict__ pb,
    const float* __restrict__ x, float* __restrict__ cls) {
  using namespace rb;
  const int bt = blockIdx.x, b = bt / T, t = bt % T, tid = threadIdx.x;
  __shared__ float cr[C];
  for (int i = tid; i < C; i += 256) cr[i] = ce[(long)bt * C + i];
  __syncthreads();
  for (int c = tid; c < C; c += 256) {
    float acc = pb[c] + x[((long)b * N + t) * C + c];
    for (int k2 = 0; k2 < C; ++k2) acc += cr[k2] * pw[(long)k2 * C + c];
    cls[(long)bt * C + c] = acc;
  }
}

// Generic per-row layernorm for the 168 cls rows
__global__ __launch_bounds__(256) void rb_lnrow(
    const float* __restrict__ src, const float* __restrict__ g,
    const float* __restrict__ bb, float* __restrict__ dst) {
  using namespace rb;
  const int row = blockIdx.x, tid = threadIdx.x;
  __shared__ float r512[C];
  __shared__ float red[256];
  for (int i = tid; i < C; i += 256) r512[i] = src[(long)row * C + i];
  __syncthreads();
  float p = 0.f;
  for (int i = tid; i < C; i += 256) p += r512[i];
  red[tid] = p; __syncthreads();
  for (int s2 = 128; s2 > 0; s2 >>= 1) { if (tid < s2) red[tid] += red[tid + s2]; __syncthreads(); }
  float mean = red[0] * (1.f / C);
  __syncthreads();
  float q2 = 0.f;
  for (int i = tid; i < C; i += 256) { float dd = r512[i] - mean; q2 += dd * dd; }
  red[tid] = q2; __syncthreads();
  for (int s2 = 128; s2 > 0; s2 >>= 1) { if (tid < s2) red[tid] += red[tid + s2]; __syncthreads(); }
  float rs = rsqrtf(red[0] * (1.f / C) + EPS);
  for (int i = tid; i < C; i += 256) dst[(long)row * C + i] = (r512[i] - mean) * rs * g[i] + bb[i];
}

// K4c: h = gelu_exact(ln2 @ fc1_w + fc1_b)
__global__ __launch_bounds__(256) void rb_fc1_gelu(
    const float* __restrict__ ln2, const float* __restrict__ w,
    const float* __restrict__ bb, float* __restrict__ hbuf) {
  using namespace rb;
  const int blk = blockIdx.x, row = blk >> 3, ch = blk & 7, tid = threadIdx.x;
  __shared__ float lr[C];
  for (int i = tid; i < C; i += 256) lr[i] = ln2[(long)row * C + i];
  __syncthreads();
  const int j = ch * 256 + tid;
  float acc = bb[j];
  for (int k2 = 0; k2 < C; ++k2) acc += lr[k2] * w[(long)k2 * HID + j];
  hbuf[(long)row * HID + j] = 0.5f * acc * (1.f + erff(acc * 0.70710678118654752f));
}

// K4d: cls += h @ fc2_w + fc2_b
__global__ __launch_bounds__(256) void rb_fc2_res(
    const float* __restrict__ hbuf, const float* __restrict__ w,
    const float* __restrict__ bb, float* __restrict__ cls) {
  using namespace rb;
  const int row = blockIdx.x, tid = threadIdx.x;
  __shared__ float hr[HID];
  for (int i = tid; i < HID; i += 256) hr[i] = hbuf[(long)row * HID + i];
  __syncthreads();
  for (int c = tid; c < C; c += 256) {
    float acc = bb[c] + cls[(long)row * C + c];
    for (int k2 = 0; k2 < HID; ++k2) acc += hr[k2] * w[(long)k2 * C + c];
    cls[(long)row * C + c] = acc;
  }
}

// K5b: overwrite k1 rows n<T with ln3(cls) @ kv1_w[:, :512]
__global__ __launch_bounds__(256) void rb_k1fix(
    const float* __restrict__ lncls, const float* __restrict__ kv1w, unsigned short* __restrict__ k1) {
  using namespace rb;
  const int bt = blockIdx.x, b = bt / T, t = bt % T, tid = threadIdx.x;
  __shared__ float lr[C];
  for (int i = tid; i < C; i += 256) lr[i] = lncls[(long)bt * C + i];
  __syncthreads();
  for (int c = tid; c < C; c += 256) {
    float acc = 0.f;
    for (int k2 = 0; k2 < C; ++k2) acc += lr[k2] * kv1w[(long)k2 * 1024 + c];
    int h = c >> 6, d = c & 63;
    k1[(((long)b * H + h) * N + t) * 64 + d] = f2bf(acc);
  }
}

// K6: raw final scores (q1 . k1^T) -> d_out, plus per-chunk (max, sumexp) stats.
// k-tile staged by the Tensor Data Mover when available (2D D#, OOB rows -> 0).
__global__ __launch_bounds__(256) void rb_scores(
    const float* __restrict__ q1, const unsigned short* __restrict__ k1,
    float* __restrict__ out, float* __restrict__ cstats) {
  using namespace rb;
  __shared__ unsigned short kt[256 * 64];
  __shared__ float wred[8 * 32];
  __shared__ float wsum[8 * 32];
  const int tid = threadIdx.x, lane = tid & 31, w = tid >> 5;
  const int half = lane >> 4, nn = lane & 15;
  const int chunk = blockIdx.x, bh = blockIdx.y;
  const int nBase = chunk * 256;
  const float NEGINF = -__builtin_inff();

#if RB_TDM
  if (w == 0) {
    unsigned ldsAddr = (unsigned)(unsigned long long)(void*)kt;
    unsigned long long ga =
        (unsigned long long)(const void*)(k1 + (long)bh * N * 64 + (long)nBase * 64);
    // D# group0: count=1, lds_addr, 57-bit global addr, type=2
    v4u g0 = { 1u, ldsAddr, (unsigned)ga,
               (unsigned)((ga >> 32) & 0x1ffffffull) | (2u << 30) };
    unsigned td1 = (unsigned)(N - nBase);  // rows available; OOB rows read as zero
    // D# group1: data_size=2B; tensor_dim0=64; tensor_dim1=td1; tile=64x256; stride0=64
    v8i g1 = { (int)(1u << 16), (int)(64u << 16), (int)((td1 & 0xffffu) << 16),
               (int)((td1 >> 16) | (64u << 16)), 256, 64, 0, 0 };
    v4i g2 = {0, 0, 0, 0};
    v4i g3 = {0, 0, 0, 0};
    v8i g4 = {0, 0, 0, 0, 0, 0, 0, 0};
    __builtin_amdgcn_tensor_load_to_lds(g0, g1, g2, g3, g4, 0);
    RB_WAIT_TENSOR();
  }
#else
  {
    const int n = nBase + tid;
    const bool val = n < N;
    uint4 z; z.x = z.y = z.z = z.w = 0u;
    uint4* kd = (uint4*)(kt + tid * 64);
    const uint4* kp = (const uint4*)(k1 + (long)bh * N * 64 + (long)n * 64);
#if RB_ASYNC
    if (val) {
#pragma unroll
      for (int jj = 0; jj < 8; ++jj) asyncLoad16(kp + jj, kd + jj);
    } else {
#pragma unroll
      for (int jj = 0; jj < 8; ++jj) kd[jj] = z;
    }
    RB_WAIT_ASYNC();
#else
#pragma unroll
    for (int jj = 0; jj < 8; ++jj) kd[jj] = val ? kp[jj] : z;
#endif
  }
#endif
  v16bf qf[2][2];
#pragma unroll
  for (int mt = 0; mt < 2; ++mt)
#pragma unroll
    for (int ks = 0; ks < 2; ++ks)
      qf[mt][ks] = qFrag(q1 + ((long)bh * 32 + mt * 16 + nn) * 64, half, ks);
  __syncthreads();

  const int nOff = w * 32;
  v8f S[2][2];
#pragma unroll
  for (int mt = 0; mt < 2; ++mt)
#pragma unroll
    for (int sub = 0; sub < 2; ++sub)
#pragma unroll
      for (int r = 0; r < 8; ++r) S[mt][sub][r] = 0.f;
#pragma unroll
  for (int sub = 0; sub < 2; ++sub)
#pragma unroll
    for (int ks = 0; ks < 2; ++ks) {
      v16bf bf = bFrag(kt, (size_t)(nOff + sub * 16 + nn), 64, half, ks * 32);
      S[0][sub] = wmma_bf16(qf[0][ks], bf, S[0][sub]);
      S[1][sub] = wmma_bf16(qf[1][ks], bf, S[1][sub]);
    }
  // store raw scores (unnormalized)
#pragma unroll
  for (int mt = 0; mt < 2; ++mt)
#pragma unroll
    for (int sub = 0; sub < 2; ++sub)
#pragma unroll
      for (int r = 0; r < 8; ++r) {
        int t = mt * 16 + r + 8 * half;
        int n = nBase + nOff + sub * 16 + nn;
        if (t < T && n < N) out[((long)bh * T + t) * N + n] = S[mt][sub][r];
      }
  // mask invalid cols for stats
#pragma unroll
  for (int sub = 0; sub < 2; ++sub) {
    if (nBase + nOff + sub * 16 + nn >= N) {
#pragma unroll
      for (int mt = 0; mt < 2; ++mt)
#pragma unroll
        for (int r = 0; r < 8; ++r) S[mt][sub][r] = NEGINF;
    }
  }
  float rm[2][8];
#pragma unroll
  for (int mt = 0; mt < 2; ++mt)
#pragma unroll
    for (int r = 0; r < 8; ++r) rm[mt][r] = redMax16(fmaxf(S[mt][0][r], S[mt][1][r]));
  if (nn == 0) {
#pragma unroll
    for (int mt = 0; mt < 2; ++mt)
#pragma unroll
      for (int r = 0; r < 8; ++r) wred[w * 32 + mt * 16 + r + 8 * half] = rm[mt][r];
  }
  __syncthreads();
#pragma unroll
  for (int mt = 0; mt < 2; ++mt)
#pragma unroll
    for (int r = 0; r < 8; ++r) {
      int t = mt * 16 + r + 8 * half;
      float m = NEGINF;
      for (int w2 = 0; w2 < 8; ++w2) m = fmaxf(m, wred[w2 * 32 + t]);
      float a0 = (S[mt][0][r] == NEGINF) ? 0.f : __expf(S[mt][0][r] - m);
      float a1 = (S[mt][1][r] == NEGINF) ? 0.f : __expf(S[mt][1][r] - m);
      float ls = redSum16(a0 + a1);
      if (nn == 0) wsum[w * 32 + t] = ls;
    }
  __syncthreads();
  if (tid < 32) {
    int t = tid;
    float M = NEGINF, SS = 0.f;
    for (int w2 = 0; w2 < 8; ++w2) { M = fmaxf(M, wred[w2 * 32 + t]); SS += wsum[w2 * 32 + t]; }
    long o = ((long)bh * NTCH + chunk) * 64 + t * 2;
    cstats[o] = M; cstats[o + 1] = SS;
  }
}

// K7: combine chunk stats -> final (M, L) per (b,h,t)
__global__ void rb_stats(const float* __restrict__ cstats, float* __restrict__ fml) {
  using namespace rb;
  const int bh = blockIdx.x, t = threadIdx.x;
  if (t >= T) return;
  float M = -__builtin_inff();
  for (int c2 = 0; c2 < NTCH; ++c2) M = fmaxf(M, cstats[((long)bh * NTCH + c2) * 64 + t * 2]);
  float L = 0.f;
  for (int c2 = 0; c2 < NTCH; ++c2) {
    float m = cstats[((long)bh * NTCH + c2) * 64 + t * 2];
    float s = cstats[((long)bh * NTCH + c2) * 64 + t * 2 + 1];
    L += s * __expf(m - M);
  }
  long o = ((long)bh * T + t) * 2;
  fml[o] = M; fml[o + 1] = (L > 0.f) ? L : 1.f;
}

// K8: normalize scores in place -> softmax
__global__ __launch_bounds__(256) void rb_norm(float* __restrict__ out, const float* __restrict__ fml) {
  using namespace rb;
  const long TOT = (long)B * H * T * N;
  const long base = (long)blockIdx.x * 1024 + threadIdx.x;
#pragma unroll
  for (int k2 = 0; k2 < 4; ++k2) {
    long i = base + (long)k2 * 256;
    if (i < TOT) {
      long r = i / N;
      out[i] = __expf(out[i] - fml[r * 2]) / fml[r * 2 + 1];
    }
  }
}

extern "C" void kernel_launch(void* const* d_in, const int* in_sizes, int n_in,
                              void* d_out, int out_size, void* d_ws, size_t ws_size,
                              hipStream_t stream) {
  using namespace rb;
  (void)in_sizes; (void)n_in; (void)out_size; (void)ws_size;
  const float* x       = (const float*)d_in[0];
  const float* ln1_g   = (const float*)d_in[1];
  const float* ln1_b   = (const float*)d_in[2];
  const float* kv0_w   = (const float*)d_in[3];
  const float* q0_w    = (const float*)d_in[4];
  const float* proj0_w = (const float*)d_in[5];
  const float* proj0_b = (const float*)d_in[6];
  const float* ln2_g   = (const float*)d_in[7];
  const float* ln2_b   = (const float*)d_in[8];
  const float* fc1_w   = (const float*)d_in[9];
  const float* fc1_b   = (const float*)d_in[10];
  const float* fc2_w   = (const float*)d_in[11];
  const float* fc2_b   = (const float*)d_in[12];
  const float* ln3_g   = (const float*)d_in[13];
  const float* ln3_b   = (const float*)d_in[14];
  const float* kv1_w   = (const float*)d_in[15];
  const float* q1_w    = (const float*)d_in[16];
  float* out = (float*)d_out;

  char* ws = (char*)d_ws;
  size_t off = 0;
  auto alloc = [&](size_t bytes) -> void* {
    void* p = ws + off;
    off = (off + bytes + 255) & ~(size_t)255;
    return p;
  };
  unsigned short* k0   = (unsigned short*)alloc((size_t)B * H * N * D * 2);
  unsigned short* v0   = (unsigned short*)alloc((size_t)B * H * N * D * 2);
  unsigned short* k1b  = (unsigned short*)alloc((size_t)B * H * N * D * 2);
  unsigned short* wt0  = (unsigned short*)alloc((size_t)1024 * C * 2);
  unsigned short* wt1  = (unsigned short*)alloc((size_t)512 * C * 2);
  float* bias0  = (float*)alloc(1024 * 4);
  float* bias1  = (float*)alloc(512 * 4);
  float* q0     = (float*)alloc((size_t)B * H * 32 * 64 * 4);
  float* q1     = (float*)alloc((size_t)B * H * 32 * 64 * 4);
  float* pO     = (float*)alloc((size_t)B * H * 64 * 32 * 64 * 4);
  float* pML    = (float*)alloc((size_t)B * H * 64 * 64 * 4);
  float* ce     = (float*)alloc((size_t)B * T * C * 4);
  float* cls    = (float*)alloc((size_t)B * T * C * 4);
  float* ln2row = (float*)alloc((size_t)B * T * C * 4);
  float* lncls  = (float*)alloc((size_t)B * T * C * 4);
  float* hbuf   = (float*)alloc((size_t)B * T * HID * 4);
  float* cstats = (float*)alloc((size_t)B * H * NTCH * 64 * 4);
  float* fml    = (float*)alloc((size_t)B * H * T * 2 * 4);

  const float scale = 0.125f;  // D^-0.5

  rb_prep_w<<<1536, 128, 0, stream>>>(kv0_w, kv1_w, ln1_g, ln1_b, ln3_g, ln3_b,
                                      wt0, bias0, wt1, bias1);
  rb_ln_dualgemm<<<(TOTROWS + 127) / 128, 256, 0, stream>>>(x, wt0, bias0, wt1, bias1,
                                                            k0, v0, k1b);
  rb_compute_q<<<B, 256, 0, stream>>>(x, (long)N * C, ln1_g, ln1_b, q0_w, q0, scale);
  rb_attn0<<<B * H * NSPLIT, 256, 0, stream>>>(q0, k0, v0, pO, pML);
  rb_attn0_combine<<<B * H, 256, 0, stream>>>(pO, pML, ce);
  rb_proj_res<<<B * T, 256, 0, stream>>>(ce, proj0_w, proj0_b, x, cls);
  rb_lnrow<<<B * T, 256, 0, stream>>>(cls, ln2_g, ln2_b, ln2row);
  rb_fc1_gelu<<<B * T * 8, 256, 0, stream>>>(ln2row, fc1_w, fc1_b, hbuf);
  rb_fc2_res<<<B * T, 256, 0, stream>>>(hbuf, fc2_w, fc2_b, cls);
  rb_lnrow<<<B * T, 256, 0, stream>>>(cls, ln3_g, ln3_b, lncls);
  rb_k1fix<<<B * T, 256, 0, stream>>>(lncls, kv1_w, k1b);
  rb_compute_q<<<B, 256, 0, stream>>>(cls, (long)T * C, ln3_g, ln3_b, q1_w, q1, scale);
  rb_scores<<<dim3(NTCH, B * H), 256, 0, stream>>>(q1, k1b, out, cstats);
  rb_stats<<<B * H, 32, 0, stream>>>(cstats, fml);
  const long TOT = (long)B * H * T * N;
  rb_norm<<<(unsigned)((TOT + 1023) / 1024), 256, 0, stream>>>(out, fml);
}
